// CA_48919677501495
// MI455X (gfx1250) — matrix operand order repaired and checked by reading, not compiled, not run on hardware
//
#include <hip/hip_runtime.h>
#include <hip/hip_bf16.h>

// ---------- WMMA fragment helpers (gfx1250, wave32) ----------
typedef __attribute__((ext_vector_type(16))) _Float16 v16h;
typedef __attribute__((ext_vector_type(8)))  _Float16 v8h;
typedef __attribute__((ext_vector_type(8)))  float    v8f;

union HFrag { v16h f; v8h h[2]; };

// A fragment (16x32) from a per-lane base pointer + constant element offset.
// Per-lane base already encodes: row (lane&15)*ld and klo=(lane>>4)*8.
__device__ __forceinline__ v16h load_frA(const _Float16* p, int off) {
  HFrag fr;
  fr.h[0] = *(const v8h*)(p + off);
  fr.h[1] = *(const v8h*)(p + off + 16);
  return fr.f;
}

// B fragment (32x16) from a per-lane base pointer + constant element offset.
// Per-lane base already encodes: col (lane&15)*ld and klo=(lane>>4)*8.
__device__ __forceinline__ v16h load_frB(const _Float16* p, int off) {
  HFrag fr;
  fr.h[0] = *(const v8h*)(p + off);
  fr.h[1] = *(const v8h*)(p + off + 16);
  return fr.f;
}

__device__ __forceinline__ v8f wmma16(v16h a, v16h b, v8f c) {
  return __builtin_amdgcn_wmma_f32_16x16x32_f16(false, a, false, b, (short)0, c,
                                                false, false);
}

// ---------- problem constants ----------
#define BB    8
#define CC    256
#define HH    32
#define WW    32
#define NN    1024     // H*W
#define GG    8        // heads
#define DD    64       // head_dim
#define INNER 512      // GG*DD
#define CO3   1536     // 3*INNER
#define KTAP  9
#define HP    34       // padded spatial (1-pixel halo)

// =====================================================================
// Pack kernels: f32 -> f16 with WMMA-friendly layouts
// =====================================================================
__global__ void pack_w_kernel(const float* __restrict__ w, _Float16* __restrict__ Wp) {
  int i = blockIdx.x * blockDim.x + threadIdx.x;           // over CO3*CC*9
  if (i >= CO3 * CC * KTAP) return;
  int tap = i % KTAP;
  int ci  = (i / KTAP) % CC;
  int co  = i / (KTAP * CC);
  Wp[tap * CO3 * CC + co * CC + ci] = (_Float16)w[i];
}

// Zero-padded, channel-last activation tensor: Xp[b][hp][wp][ci], hp/wp in [0,34).
// Interior (1..32) holds x; 1-pixel halo is zero -> conv taps need NO masking.
__global__ void pack_xp_kernel(const float* __restrict__ x, _Float16* __restrict__ Xp) {
  int i = blockIdx.x * blockDim.x + threadIdx.x;           // over B*34*34*CC
  if (i >= BB * HP * HP * CC) return;
  int ci = i % CC;
  int wp = (i / CC) % HP;
  int hp = (i / (CC * HP)) % HP;
  int b  = i / (CC * HP * HP);
  float v = 0.f;
  if (hp >= 1 && hp <= HH && wp >= 1 && wp <= WW) {
    int n = (hp - 1) * WW + (wp - 1);
    v = x[((size_t)b * CC + ci) * NN + n];
  }
  Xp[i] = (_Float16)v;
}

__global__ void pack_wo_kernel(const float* __restrict__ w, _Float16* __restrict__ Wo) {
  int i = blockIdx.x * blockDim.x + threadIdx.x;           // over C*INNER
  if (i >= CC * INNER) return;
  Wo[i] = (_Float16)w[i];                                   // already [c][i] row-major
}

// =====================================================================
// qkv 3x3 conv as 9 shifted 1x1 GEMMs over the zero-padded tensor.
// Grid: (CO3/128, NN/128, B), 256 threads (8 waves). Wave owns 16 co rows.
// Per-lane base pointers precomputed once; (tap,k0) fold into VMEM immediates.
// Writes Q,K row-major [b,g][n][d] f16 (packed b128) and V^T [b,g][d][n] f16.
// =====================================================================
__global__ __launch_bounds__(256)
void qkv_conv_kernel(const _Float16* __restrict__ Wp, const _Float16* __restrict__ Xp,
                     _Float16* __restrict__ Qh, _Float16* __restrict__ Kh,
                     _Float16* __restrict__ Vth) {
  const int w     = threadIdx.x >> 5;
  const int lane  = threadIdx.x & 31;
  const int lo    = lane & 15;
  const int hi8   = (lane >> 4) << 3;
  const int klo16 = (lane >> 4) << 4;
  const int cw0   = blockIdx.x * 128 + w * 16;
  const int n0    = blockIdx.y * 128;
  const int b     = blockIdx.z;

  const _Float16* __restrict__ XpB = Xp  + (size_t)b * HP * HP * CC;
  _Float16* __restrict__ QhB = Qh  + (size_t)b * GG * NN * DD;
  _Float16* __restrict__ KhB = Kh  + (size_t)b * GG * NN * DD;
  _Float16* __restrict__ VtB = Vth + (size_t)b * GG * DD * NN;

  // per-lane base pointers (computed once)
  const _Float16* pW = Wp + (cw0 + lo) * CC + hi8;      // A: weights
  const _Float16* pX[8];                                // B: activations per tile
#pragma unroll
  for (int t = 0; t < 8; ++t) {
    int ncol = n0 + t * 16 + lo;
    int h = ncol >> 5, x = ncol & 31;
    pX[t] = XpB + ((h + 1) * HP + (x + 1)) * CC + hi8;
  }

  const v8f zero8 = {};
  v8f acc[8];
#pragma unroll
  for (int t = 0; t < 8; ++t) acc[t] = zero8;

#pragma unroll
  for (int tap = 0; tap < KTAP; ++tap) {
    const int dy = tap / 3 - 1;
    const int dx = tap % 3 - 1;
    const int doff = (dy * HP + dx) * CC;     // compile-time per tap
    const int wtap = tap * CO3 * CC;          // compile-time per tap (<8MB imm)
#pragma unroll
    for (int k0 = 0; k0 < CC; k0 += 32) {
      v16h a = load_frA(pW, wtap + k0);
      v16h bfr[8];
#pragma unroll
      for (int t = 0; t < 8; ++t) bfr[t] = load_frB(pX[t], doff + k0);
#pragma unroll
      for (int t = 0; t < 8; ++t) acc[t] = wmma16(a, bfr[t], acc[t]);
    }
  }

  // Epilogue: Q/K pack 8 consecutive dd into one b128 store; V^T scalar scatter.
  const int co0r = cw0 + hi8;          // first of 8 consecutive co for this lane-half
  const int sec  = co0r >> 9;          // 0=q, 1=k, 2=v (uniform per wave)
  const int rem  = co0r & 511;
  const int g    = rem >> 6;
  const int dd0  = rem & 63;           // multiple of 8
#pragma unroll
  for (int t = 0; t < 8; ++t) {
    int ncol = n0 + t * 16 + lo;
    v8h pk;
#pragma unroll
    for (int j = 0; j < 8; ++j) pk[j] = (_Float16)acc[t][j];
    if (sec == 0) {
      *(v8h*)(QhB + g * NN * DD + ncol * DD + dd0) = pk;
    } else if (sec == 1) {
      *(v8h*)(KhB + g * NN * DD + ncol * DD + dd0) = pk;
    } else {
#pragma unroll
      for (int j = 0; j < 8; ++j)
        VtB[g * DD * NN + (dd0 + j) * NN + ncol] = pk[j];
    }
  }
}

// =====================================================================
// Row norms: qn/kn[b,g,n] = sqrt(sum_d q^2 + 1e-4)
// =====================================================================
__global__ void norms_kernel(const _Float16* __restrict__ Qh,
                             const _Float16* __restrict__ Kh,
                             float* __restrict__ qn, float* __restrict__ kn) {
  int idx = blockIdx.x * blockDim.x + threadIdx.x;     // over B*G*N
  if (idx >= BB * GG * NN) return;
  const v8h* q8 = (const v8h*)(Qh + (size_t)idx * DD);
  const v8h* k8 = (const v8h*)(Kh + (size_t)idx * DD);
  float sq = 0.f, sk = 0.f;
#pragma unroll
  for (int c = 0; c < 8; ++c) {
    v8h qv = q8[c], kv = k8[c];
#pragma unroll
    for (int j = 0; j < 8; ++j) {
      float a = (float)qv[j]; sq += a * a;
      float d = (float)kv[j]; sk += d * d;
    }
  }
  qn[idx] = sqrtf(sq + 1e-4f);
  kn[idx] = sqrtf(sk + 1e-4f);
}

// =====================================================================
// Fused attention: per (b, g, 128-row block).
//   S = Q K^T (WMMA) -> cosine norm + separable Gaussian pos-bias ->
//   f16 S tile via LDS -> O += S V (WMMA). Writes AO^T[b][n][g*64+dd] f16.
// Grid: (NN/128, G, B), 256 threads (8 waves, wave owns 16 n rows).
// Wave-uniform y-row + t-parity structure makes the bias lookups cheap:
//   ty: one uniform LDS read per (mb,t); tx: 2x8 per-lane table built once.
// =====================================================================
__global__ __launch_bounds__(256)
void attn_kernel(const _Float16* __restrict__ Qh, const _Float16* __restrict__ Kh,
                 const _Float16* __restrict__ Vth, const float* __restrict__ qn,
                 const float* __restrict__ kn, const float* __restrict__ hs,
                 _Float16* __restrict__ AOt) {
  __shared__ float T[32];
  __shared__ float Sy[32];
  __shared__ _Float16 Sh[8][16][136];   // per-wave 16x128 S tile (+pad)

  const int g = blockIdx.y, b = blockIdx.z;
  const int nbase = blockIdx.x * 128;
  const int tid = threadIdx.x;

  if (tid < 32) {
    float s  = 1.f / (1.f + expf(-hs[g])) * (0.4f - 0.003f) + 0.003f;
    float sf = 1.f / (2.f * s * s);
    T[tid] = expf(-sf * (float)(tid * tid) * (1.f / 1024.f));  // (d/32)^2
  }
  __syncthreads();
  if (tid < 32) {
    float acc = 0.f;
    for (int j = 0; j < 32; ++j) { int d = tid - j; d = d < 0 ? -d : d; acc += T[d]; }
    Sy[tid] = acc;
  }
  __syncthreads();

  const int w = tid >> 5, lane = tid & 31;
  const int lo = lane & 15, hi8 = (lane >> 4) << 3;
  const int klo16 = (lane >> 4) << 4;
  const size_t bg = (size_t)b * GG + g;
  const _Float16* __restrict__ QhB = Qh  + bg * NN * DD;
  const _Float16* __restrict__ KhB = Kh  + bg * NN * DD;
  const _Float16* __restrict__ VtB = Vth + bg * DD * NN;
  const float*    __restrict__ qnB = qn + bg * NN;
  const float*    __restrict__ knB = kn + bg * NN;
  _Float16* __restrict__ AOtB = AOt + (size_t)b * NN * INNER;

  // per-lane base pointers (computed once); all tile offsets are immediates
  const _Float16* pQ = QhB + (nbase + w * 16 + lo) * DD + hi8;   // A pattern
  const _Float16* pK = KhB + lo * DD + hi8;                      // B pattern
  const _Float16* pV = VtB + lo * NN + hi8;                      // B pattern

  v16h qf0 = load_frA(pQ, 0);
  v16h qf1 = load_frA(pQ, 32);

  // Wave's 16 rows are consecutive n starting at a multiple of 16 -> single y.
  const int yn  = (nbase + w * 16) >> 5;          // wave-uniform row y
  const int x0h = ((w & 1) << 4) + hi8;           // xn = x0h + j  (per lane-half)

  float qrow[8];
  float cp[2][8];     // T[|xn-xm|] * rsrow, for t-parity p (xm = p*16 + lo)
#pragma unroll
  for (int j = 0; j < 8; ++j) {
    int nr = nbase + w * 16 + j + hi8;
    qrow[j] = qnB[nr];
    int xn = x0h + j;
    float rs = 1.f / (Sy[yn] * Sy[xn]);
#pragma unroll
    for (int p = 0; p < 2; ++p) {
      int dxv = xn - (p << 4) - lo; dxv = dxv < 0 ? -dxv : dxv;
      cp[p][j] = T[dxv] * rs;
    }
  }

  const v8f zero8 = {};
  v8f oacc[4];
#pragma unroll
  for (int t = 0; t < 4; ++t) oacc[t] = zero8;

#pragma unroll
  for (int mb = 0; mb < 8; ++mb) {
    const int m0 = mb * 128;
    // prefetch next m-block of K and V toward the WGP (global_prefetch_b8)
    if (mb + 1 < 8) {
      int mn = m0 + 128;
      __builtin_prefetch(KhB + (mn + lane * 4) * DD, 0, 1);
      __builtin_prefetch(VtB + (lane * 2) * NN + mn, 0, 1);
    }
    v8f sacc[8];
#pragma unroll
    for (int t = 0; t < 8; ++t) sacc[t] = zero8;

    v16h bfr[8];
#pragma unroll
    for (int t = 0; t < 8; ++t) bfr[t] = load_frB(pK, (m0 + t * 16) * DD);
#pragma unroll
    for (int t = 0; t < 8; ++t) sacc[t] = wmma16(qf0, bfr[t], sacc[t]);
#pragma unroll
    for (int t = 0; t < 8; ++t) bfr[t] = load_frB(pK, (m0 + t * 16) * DD + 32);
#pragma unroll
    for (int t = 0; t < 8; ++t) sacc[t] = wmma16(qf1, bfr[t], sacc[t]);

    // elementwise: cosine norm + positional bias; write f16 S tile to LDS
#pragma unroll
    for (int t = 0; t < 8; ++t) {
      const int p = t & 1;
      int ymu = (mb << 2) + (t >> 1);             // wave-uniform m-row y
      int dyv = yn - ymu; dyv = dyv < 0 ? -dyv : dyv;
      float ty = T[dyv];                          // uniform LDS read
      float knv = knB[m0 + t * 16 + lo];
#pragma unroll
      for (int j = 0; j < 8; ++j) {
        float r = __builtin_amdgcn_rcpf(qrow[j] * knv + 1e-4f);  // v_rcp_f32
        float val = sacc[t][j] * (ty * cp[p][j]) * r;
        Sh[w][j + hi8][t * 16 + lo] = (_Float16)val;
      }
    }
    // O += S * V  (A frags re-read from own-wave LDS region; no barrier needed)
#pragma unroll
    for (int ks = 0; ks < 4; ++ks) {
      const _Float16* pS = &Sh[w][lo][ks * 32 + hi8];
      HFrag af;
      af.h[0] = *(const v8h*)(pS);
      af.h[1] = *(const v8h*)(pS + 16);
      v16h vfr[4];
#pragma unroll
      for (int td = 0; td < 4; ++td)
        vfr[td] = load_frB(pV, (td * 16) * NN + m0 + ks * 32);
#pragma unroll
      for (int td = 0; td < 4; ++td) oacc[td] = wmma16(af.f, vfr[td], oacc[td]);
    }
  }
  // write AO^T[b][n][g*64+dd]
#pragma unroll
  for (int td = 0; td < 4; ++td) {
    int ddv = g * DD + td * 16 + lo;
#pragma unroll
    for (int j = 0; j < 8; ++j) {
      int nr = nbase + w * 16 + j + hi8;
      AOtB[nr * INNER + ddv] = (_Float16)oacc[td][j];
    }
  }
}

// =====================================================================
// 1x1 output projection GEMM (WMMA): Pout[b][c][n] = sum_i Wo[c][i]*AO[b][i][n]
// Grid: (NN/128, CC/128, B), 256 threads.
// =====================================================================
__global__ __launch_bounds__(256)
void proj_kernel(const _Float16* __restrict__ Wo, const _Float16* __restrict__ AOt,
                 float* __restrict__ Pout) {
  const int w = threadIdx.x >> 5, lane = threadIdx.x & 31;
  const int lo = lane & 15, hi8 = (lane >> 4) << 3;
  const int klo16 = (lane >> 4) << 4;
  const int co0 = blockIdx.y * 128 + w * 16;
  const int n0  = blockIdx.x * 128;
  const int b   = blockIdx.z;
  const _Float16* __restrict__ AOtB = AOt + (size_t)b * NN * INNER;
  float* __restrict__ PoutB = Pout + (size_t)b * CC * NN;

  const _Float16* pA = Wo + (co0 + lo) * INNER + hi8;         // A pattern
  const _Float16* pB = AOtB + (n0 + lo) * INNER + hi8;        // B pattern

  const v8f zero8 = {};
  v8f acc[8];
#pragma unroll
  for (int t = 0; t < 8; ++t) acc[t] = zero8;

#pragma unroll 4
  for (int k0 = 0; k0 < INNER; k0 += 32) {
    v16h af = load_frA(pA, k0);
    v16h bfr[8];
#pragma unroll
    for (int t = 0; t < 8; ++t) bfr[t] = load_frB(pB, (t * 16) * INNER + k0);
#pragma unroll
    for (int t = 0; t < 8; ++t) acc[t] = wmma16(af, bfr[t], acc[t]);
  }
#pragma unroll
  for (int t = 0; t < 8; ++t) {
    int ncol = n0 + t * 16 + lo;
#pragma unroll
    for (int j = 0; j < 8; ++j) {
      int c = co0 + j + hi8;
      PoutB[c * NN + ncol] = acc[t][j];
    }
  }
}

// =====================================================================
// BatchNorm statistics (train mode, per-channel over B*H*W) -> scale/bias
// =====================================================================
__global__ __launch_bounds__(256)
void bnstat_kernel(const float* __restrict__ Pout, const float* __restrict__ gamma,
                   const float* __restrict__ beta, float* __restrict__ scl,
                   float* __restrict__ bias) {
  const int c = blockIdx.x, tid = threadIdx.x;
  float s = 0.f, s2 = 0.f;
  for (int i = tid; i < BB * NN; i += 256) {
    int b = i >> 10, n = i & 1023;
    float v = Pout[((size_t)b * CC + c) * NN + n];
    s += v; s2 += v * v;
  }
  __shared__ float rs[256], rs2[256];
  rs[tid] = s; rs2[tid] = s2;
  __syncthreads();
  for (int o = 128; o > 0; o >>= 1) {
    if (tid < o) { rs[tid] += rs[tid + o]; rs2[tid] += rs2[tid + o]; }
    __syncthreads();
  }
  if (tid == 0) {
    float mean = rs[0] * (1.f / (BB * NN));
    float var  = rs2[0] * (1.f / (BB * NN)) - mean * mean;
    float sc = gamma[c] * rsqrtf(var + 1e-5f);
    scl[c] = sc;
    bias[c] = beta[c] - mean * sc;
  }
}

__global__ void finalize_kernel(const float* __restrict__ Pout,
                                const float* __restrict__ scl,
                                const float* __restrict__ bias,
                                float* __restrict__ out) {
  int i = blockIdx.x * blockDim.x + threadIdx.x;     // over B*C*N
  if (i >= BB * CC * NN) return;
  int c = (i >> 10) & (CC - 1);
  float v = Pout[i] * scl[c] + bias[c];
  out[i] = v > 0.f ? v : 0.f;
}

// =====================================================================
// Host launcher
// =====================================================================
extern "C" void kernel_launch(void* const* d_in, const int* in_sizes, int n_in,
                              void* d_out, int out_size, void* d_ws, size_t ws_size,
                              hipStream_t stream) {
  const float* x     = (const float*)d_in[0];
  const float* w_qkv = (const float*)d_in[1];
  const float* hs    = (const float*)d_in[2];
  const float* w_out = (const float*)d_in[3];
  const float* gamma = (const float*)d_in[4];
  const float* beta  = (const float*)d_in[5];
  float* out = (float*)d_out;

  char* ws = (char*)d_ws;
  size_t off = 0;
  auto alloc = [&](size_t bytes) -> char* {
    char* p = ws + off;
    off = (off + bytes + 255) & ~(size_t)255;
    return p;
  };
  _Float16* Wp  = (_Float16*)alloc((size_t)KTAP * CO3 * CC * 2);
  _Float16* Xp  = (_Float16*)alloc((size_t)BB * HP * HP * CC * 2);
  _Float16* Qh  = (_Float16*)alloc((size_t)BB * GG * NN * DD * 2);
  _Float16* Kh  = (_Float16*)alloc((size_t)BB * GG * NN * DD * 2);
  _Float16* Vth = (_Float16*)alloc((size_t)BB * GG * DD * NN * 2);
  float*    qn  = (float*)alloc((size_t)BB * GG * NN * 4);
  float*    kn  = (float*)alloc((size_t)BB * GG * NN * 4);
  _Float16* AOt = (_Float16*)alloc((size_t)BB * NN * INNER * 2);
  _Float16* Wo  = (_Float16*)alloc((size_t)CC * INNER * 2);
  float*    Pout = (float*)alloc((size_t)BB * CC * NN * 4);
  float*    scl  = (float*)alloc(CC * 4);
  float*    bias = (float*)alloc(CC * 4);

  {
    int n = CO3 * CC * KTAP;
    pack_w_kernel<<<(n + 255) / 256, 256, 0, stream>>>(w_qkv, Wp);
  }
  {
    int n = BB * HP * HP * CC;
    pack_xp_kernel<<<(n + 255) / 256, 256, 0, stream>>>(x, Xp);
  }
  {
    int n = CC * INNER;
    pack_wo_kernel<<<(n + 255) / 256, 256, 0, stream>>>(w_out, Wo);
  }
  qkv_conv_kernel<<<dim3(CO3 / 128, NN / 128, BB), 256, 0, stream>>>(Wp, Xp, Qh, Kh, Vth);
  {
    int n = BB * GG * NN;
    norms_kernel<<<(n + 255) / 256, 256, 0, stream>>>(Qh, Kh, qn, kn);
  }
  attn_kernel<<<dim3(NN / 128, GG, BB), 256, 0, stream>>>(Qh, Kh, Vth, qn, kn, hs, AOt);
  proj_kernel<<<dim3(NN / 128, CC / 128, BB), 256, 0, stream>>>(Wo, AOt, Pout);
  bnstat_kernel<<<CC, 256, 0, stream>>>(Pout, gamma, beta, scl, bias);
  {
    int n = BB * CC * NN;
    finalize_kernel<<<(n + 255) / 256, 256, 0, stream>>>(Pout, scl, bias, out);
  }
  (void)in_sizes; (void)n_in; (void)out_size; (void)ws_size;
}
